// ScaledDotProductAttention_26482768347411
// MI455X (gfx1250) — matrix operand order repaired and checked by reading, not compiled, not run on hardware
//
#include <hip/hip_runtime.h>

typedef __attribute__((ext_vector_type(16))) _Float16 v16h;
typedef __attribute__((ext_vector_type(8)))  float    v8f;
typedef __attribute__((ext_vector_type(4)))  float    f32x4;   // native vector: OK for nontemporal builtins

#define S_LEN   2048
#define DK      64
#define NWAVE   4
#define SROW    2052                     // 2048 + 4 pad floats: rows 0 and 8 land on different banks
#define CTX_ELEMS (2L * 16 * 2048 * 64)  // context elements preceding attn in d_out

__global__ __launch_bounds__(128)
void attn_fused_kernel(const float* __restrict__ Qp,
                       const float* __restrict__ Kp,
                       const float* __restrict__ Vp,
                       float* __restrict__ ctx_out,
                       float* __restrict__ attn_out)
{
    extern __shared__ float smem[];
    float* sc   = smem;                 // [16][SROW] fp32 scores -> probabilities
    float* cbuf = smem + 16 * SROW;     // [NWAVE][16][64] partial contexts

    const int qt   = blockIdx.x;        // query tile 0..127
    const int bh   = blockIdx.y;        // fused batch*head 0..31
    const int tid  = threadIdx.x;
    const int lane = tid & 31;
    const int wid  = tid >> 5;

    const int L    = (qt + 1) * 16;     // valid causal row length (multiple of 16)
    const int Lpad = (L + 31) & ~31;    // rounded up to WMMA K-step

    const float scale = 0.125f;         // 1/sqrt(64)

    // A-operand lane mapping (16-bit A 16x32): M = lane&15, half = lane>>4
    const int aM = lane & 15;
    const int ah = lane >> 4;
    const int bN = lane & 15;           // B/C/D column = lane&15

    // ---------------- Phase 1: load Q tile as WMMA A operands ----------------
    v16h a0, a1;
    {
        const float* qrow = Qp + ((long)bh * S_LEN + (qt * 16 + aM)) * DK;
        #pragma unroll
        for (int j = 0; j < 8; ++j) {
            a0[j]     = (_Float16)qrow[      ah * 16 + j];      // d = half*16 + j
            a0[j + 8] = (_Float16)qrow[      ah * 16 + 8 + j];  // d = half*16 + 8 + j
            a1[j]     = (_Float16)qrow[32 +  ah * 16 + j];
            a1[j + 8] = (_Float16)qrow[32 +  ah * 16 + 8 + j];
        }
    }

    // ---- scores = QK^T * scale, causal-masked, into LDS (waves stride k-tiles) ----
    for (int kt = wid; kt <= qt; kt += NWAVE) {
        const float* krow = Kp + ((long)bh * S_LEN + (kt * 16 + bN)) * DK;
        v16h b0, b1;                    // B 32x16: lane half covers K=half*16 + j
        #pragma unroll
        for (int j = 0; j < 16; ++j) {
            b0[j] = (_Float16)krow[     ah * 16 + j];
            b1[j] = (_Float16)krow[32 + ah * 16 + j];
        }
        v8f c = {};
        c = __builtin_amdgcn_wmma_f32_16x16x32_f16(false, a0, false, b0, (short)0, c, false, false);
        c = __builtin_amdgcn_wmma_f32_16x16x32_f16(false, a1, false, b1, (short)0, c, false, false);

        #pragma unroll
        for (int r = 0; r < 8; ++r) {
            const int M = ah * 8 + r;   // D row
            float s = c[r] * scale;
            if (kt == qt && bN > M) s = -1e9f;   // causal mask on diagonal tile
            sc[M * SROW + kt * 16 + bN] = s;
        }
    }
    __syncthreads();

    // ---------------- Phase 2: fp32 softmax per row (b128 granularity) ----------------
    #pragma unroll
    for (int rr = 0; rr < 4; ++rr) {
        const int row = wid * 4 + rr;
        float* srow = sc + row * SROW;          // 16B-aligned: SROW % 4 == 0
        f32x4* srow4 = (f32x4*)srow;

        // row max (L is a multiple of 16, so 4-wide chunks never straddle it)
        float m = -3.0e38f;
        for (int c4 = lane; c4 * 4 < L; c4 += 32) {
            f32x4 v = srow4[c4];
            m = fmaxf(m, fmaxf(fmaxf(v.x, v.y), fmaxf(v.z, v.w)));
        }
        #pragma unroll
        for (int off = 16; off >= 1; off >>= 1) m = fmaxf(m, __shfl_xor(m, off, 32));

        // exp + row sum
        float sum = 0.f;
        for (int c4 = lane; c4 * 4 < L; c4 += 32) {
            f32x4 v = srow4[c4];
            v.x = __expf(v.x - m); v.y = __expf(v.y - m);
            v.z = __expf(v.z - m); v.w = __expf(v.w - m);
            srow4[c4] = v;
            sum += (v.x + v.y) + (v.z + v.w);
        }
        #pragma unroll
        for (int off = 16; off >= 1; off >>= 1) sum += __shfl_xor(sum, off, 32);
        const float inv = 1.0f / sum;

        // normalize in LDS + stream attn row out with non-temporal b128 stores
        f32x4* arow4 = (f32x4*)(attn_out + (((long)bh * S_LEN) + qt * 16 + row) * S_LEN);
        for (int c4 = lane; c4 < S_LEN / 4; c4 += 32) {
            const int c = c4 * 4;
            f32x4 v;
            if (c < L) {
                v = srow4[c4];
                v.x *= inv; v.y *= inv; v.z *= inv; v.w *= inv;
                srow4[c4] = v;                      // normalized P kept for PV
            } else {
                v = (f32x4){0.f, 0.f, 0.f, 0.f};
                if (c < Lpad) srow4[c4] = v;        // zero-pad PV tail tile
            }
            __builtin_nontemporal_store(v, &arow4[c4]);  // don't thrash L2 (K/V live there)
        }
    }
    __syncthreads();

    // ---------------- Phase 3: context = P @ V via WMMA (waves stride k) ----------------
    v8f acc0 = {}, acc1 = {}, acc2 = {}, acc3 = {};
    for (int k0 = wid * 32; k0 < Lpad; k0 += NWAVE * 32) {
        v16h ap;                        // P tile 16x32 from LDS, f32 -> f16
        const float* prow = sc + aM * SROW + k0;
        #pragma unroll
        for (int j = 0; j < 8; ++j) {
            ap[j]     = (_Float16)prow[      ah * 16 + j];
            ap[j + 8] = (_Float16)prow[      ah * 16 + 8 + j];
        }
        const float* vbase = Vp + ((long)bh * S_LEN + k0) * DK;
        v16h bv0, bv1, bv2, bv3;        // V 32x16 tiles for d-groups 0..3
        #pragma unroll
        for (int j = 0; j < 16; ++j) {
            const float* vr = vbase + (long)(ah * 16 + j) * DK + bN;
            bv0[j] = (_Float16)vr[0];
            bv1[j] = (_Float16)vr[16];
            bv2[j] = (_Float16)vr[32];
            bv3[j] = (_Float16)vr[48];
        }
        acc0 = __builtin_amdgcn_wmma_f32_16x16x32_f16(false, ap, false, bv0, (short)0, acc0, false, false);
        acc1 = __builtin_amdgcn_wmma_f32_16x16x32_f16(false, ap, false, bv1, (short)0, acc1, false, false);
        acc2 = __builtin_amdgcn_wmma_f32_16x16x32_f16(false, ap, false, bv2, (short)0, acc2, false, false);
        acc3 = __builtin_amdgcn_wmma_f32_16x16x32_f16(false, ap, false, bv3, (short)0, acc3, false, false);
    }

    // stash per-wave partial 16x64 context
    {
        float* cw = cbuf + wid * 16 * 64;
        #pragma unroll
        for (int r = 0; r < 8; ++r) {
            const int M = ah * 8 + r;
            cw[M * 64 +      bN] = acc0[r];
            cw[M * 64 + 16 + bN] = acc1[r];
            cw[M * 64 + 32 + bN] = acc2[r];
            cw[M * 64 + 48 + bN] = acc3[r];
        }
    }
    __syncthreads();

    // cross-wave reduction + store context
    for (int idx = tid; idx < 16 * 64; idx += 128) {
        const float s = cbuf[idx] + cbuf[1024 + idx] + cbuf[2048 + idx] + cbuf[3072 + idx];
        const int M = idx >> 6, d = idx & 63;
        __builtin_nontemporal_store(s, &ctx_out[((long)bh * S_LEN + qt * 16 + M) * DK + d]);
    }
}

extern "C" void kernel_launch(void* const* d_in, const int* in_sizes, int n_in,
                              void* d_out, int out_size, void* d_ws, size_t ws_size,
                              hipStream_t stream) {
    (void)in_sizes; (void)n_in; (void)out_size; (void)d_ws; (void)ws_size;
    const float* Q = (const float*)d_in[0];
    const float* K = (const float*)d_in[1];
    const float* V = (const float*)d_in[2];
    // d_in[3] (attn_mask) is the known causal mask; applied analytically in-kernel.

    float* ctx  = (float*)d_out;
    float* attn = (float*)d_out + CTX_ELEMS;

    const size_t smem = (16 * SROW + NWAVE * 16 * 64) * sizeof(float); // 147,712 B
    static_assert((16 * SROW + NWAVE * 16 * 64) * sizeof(float) < 320 * 1024, "LDS budget");

    // allow >64KB dynamic LDS (no-op if driver already permits it)
    (void)hipFuncSetAttribute((const void*)attn_fused_kernel,
                              hipFuncAttributeMaxDynamicSharedMemorySize, (int)smem);

    dim3 grid(S_LEN / 16, 32);   // (q-tile, batch*head)
    dim3 block(128);             // 4 waves of 32
    attn_fused_kernel<<<grid, block, smem, stream>>>(Q, K, V, ctx, attn);
}
